// Speller_17884243821256
// MI455X (gfx1250) — matrix operand order repaired and checked by reading, not compiled, not run on hardware
//
#include <hip/hip_runtime.h>
#include <cstddef>

typedef __attribute__((ext_vector_type(16))) __bf16 v16bf;
typedef __attribute__((ext_vector_type(8)))  __bf16 v8bf;
typedef __attribute__((ext_vector_type(8)))  float  v8f;
typedef __attribute__((ext_vector_type(4)))  unsigned int v4u;
typedef __attribute__((ext_vector_type(8)))  int v8i;
typedef __attribute__((ext_vector_type(4)))  int v4i;

#define S_DIM 512
#define H2D   512
#define ATTN_D 128
#define NCLS  64
#define BB    64
#define TT    512
#define NSTEP 128
#define KDIM  1024   // A width: [prev-input 512 | recurrent h 512]
#define NDIM  2048   // 4*S_DIM (gate-interleaved: n' = s*4 + gate)

__device__ __forceinline__ float sigf(float x){ return 1.0f/(1.0f+__expf(-x)); }

__device__ __forceinline__ v16bf ldA16(const __bf16* p){
  // A-fragment: two 16B chunks at k0 and k0+16 (16-bit A layout, 16x32)
  v8bf lo = *(const v8bf*)p;
  v8bf hi = *(const v8bf*)(p+16);
  return __builtin_shufflevector(lo,hi,0,1,2,3,4,5,6,7,8,9,10,11,12,13,14,15);
}

__device__ __forceinline__ v8f wmma_bf16(v16bf a, v16bf b, v8f c){
  return __builtin_amdgcn_wmma_f32_16x16x32_bf16(false, a, false, b, (short)0, c, false, false);
}

// ---------------------------------------------------------------------------
// Pack fp32 weights (out,in) into bf16 WMMA-B-fragment order with gate
// interleave permutation n' = s*4 + gate.  Layout: [nt][kt][lane][j] where
// K = kt*32 + (lane>>4)*8 + (j&7) + (j>>3)*16,  n' = nt*16 + (lane&15).
// ---------------------------------------------------------------------------
__global__ void pack_weights(const float* __restrict__ Wih,
                             const float* __restrict__ Whh,
                             __bf16* __restrict__ Wp,
                             int ihStride, int ihColOff) {
  int idx = blockIdx.x * 256 + threadIdx.x;           // 0 .. 2M-1
  int j    = idx & 15;
  int lane = (idx >> 4) & 31;
  int kt   = (idx >> 9) & 31;
  int nt   = idx >> 14;
  int K    = kt*32 + (lane>>4)*8 + (j&7) + (j>>3)*16;
  int np   = nt*16 + (lane & 15);
  int s    = np >> 2;
  int gate = np & 3;
  int n    = gate*512 + s;
  float v = (K < 512) ? Wih[(size_t)n*ihStride + ihColOff + K]
                      : Whh[(size_t)n*512 + (K - 512)];
  Wp[idx] = (__bf16)v;
}

// ---------------------------------------------------------------------------
// Init: A0[0] = [bf16(h[:,0,:]) | 0],  A1[0] = 0,  c1 = c2 = 0
// ---------------------------------------------------------------------------
__global__ void init_state(const float* __restrict__ h,
                           __bf16* __restrict__ A0_0,
                           __bf16* __restrict__ A1_0,
                           float* __restrict__ c1,
                           float* __restrict__ c2) {
  int idx = blockIdx.x * 256 + threadIdx.x;   // 0..65535 over (b, k<1024)
  int b = idx >> 10;
  int k = idx & 1023;
  A0_0[idx] = (k < 512) ? (__bf16)h[(size_t)b*TT*H2D + k] : (__bf16)0.0f;
  A1_0[idx] = (__bf16)0.0f;
  if (idx < BB*S_DIM) { c1[idx] = 0.0f; c2[idx] = 0.0f; }
}

// ---------------------------------------------------------------------------
// psi_h[b,t,a] = h[b,t,:] . Wpsi[a,:] + bpsi[a]   (one-time, 4.3 GFLOP)
// 2 rows per block; the 2x512 fp32 tile is staged into LDS by the Tensor
// Data Mover (tensor_load_to_lds + s_wait_tensorcnt), issued by wave 0.
// ---------------------------------------------------------------------------
__global__ void psi_precompute(const float* __restrict__ h,
                               const float* __restrict__ Wpsi,
                               const float* __restrict__ bpsi,
                               float* __restrict__ psi) {
  __shared__ float hrow[2][512];
  int tid = threadIdx.x;
  int r0 = blockIdx.x * 2;                     // 2 of 32768 rows

  if (tid < 32) {                              // wave 0 issues the TDM op
    unsigned long long ga = (unsigned long long)(const void*)(h + (size_t)r0*512);
    unsigned int ldsOff = (unsigned int)(unsigned long long)(void*)&hrow[0][0];
    v4u g0;
    g0[0] = 1u;                                           // count=1, user mode
    g0[1] = ldsOff;                                       // lds_addr (bytes)
    g0[2] = (unsigned int)(ga & 0xFFFFFFFFull);           // global_addr[31:0]
    g0[3] = (unsigned int)((ga >> 32) & 0x1FFFFFFull)     // global_addr[56:32]
          | (2u << 30);                                   // type = 2 (image)
    v8i g1;
    g1[0] = (int)(2u << 16);            // workgroup_mask=0, data_size=2 (4B)
    g1[1] = (int)(512u << 16);          // tensor_dim0[15:0] in bits 63:48
    g1[2] = (int)(0u | (2u << 16));     // tensor_dim0[31:16]=0, tensor_dim1 lo=2
    g1[3] = (int)(0u | (512u << 16));   // tensor_dim1 hi=0, tile_dim0=512
    g1[4] = (int)2;                     // tile_dim1=2, tile_dim2=0
    g1[5] = (int)512;                   // tensor_dim0_stride[31:0] = 512
    g1[6] = 0;                          // stride hi, tensor_dim1_stride lo
    g1[7] = 0;                          // tensor_dim1_stride hi
    v4i g2 = {0,0,0,0};
    v4i g3 = {0,0,0,0};
    v8i g4 = {0,0,0,0,0,0,0,0};
    __builtin_amdgcn_tensor_load_to_lds(g0, g1, g2, g3, g4, 0);
    __builtin_amdgcn_s_wait_tensorcnt(0);
  }
  __syncthreads();

  int lr = tid >> 7;
  int a  = tid & 127;
  const float* wr = Wpsi + (size_t)a*512;
  float acc = bpsi[a];
  #pragma unroll 4
  for (int k = 0; k < 512; ++k) acc += wr[k]*hrow[lr][k];
  psi[(size_t)(r0+lr)*ATTN_D + a] = acc;
}

// ---------------------------------------------------------------------------
// Fused LSTM step: g = A @ Wp (WMMA bf16) [+ onehot gather + biases] -> gates
// -> c,h update.  grid 32 x 256 (8 waves).  Each wave: 1 N-tile x 2 M-tiles,
// software-pipelined (prefetch kt+1 fragments before kt's WMMAs).
// Workgroup wg owns n' in [wg*64, wg*64+64)  <=>  s in [wg*16, wg*16+16).
// ---------------------------------------------------------------------------
__global__ void lstm_step(const __bf16* __restrict__ aIn,    // [64][1024] bf16
                          const __bf16* __restrict__ wp,     // packed weights
                          const float*  __restrict__ bih,
                          const float*  __restrict__ bhh,
                          const float*  __restrict__ gatherW,// Wih0 (layer0) or null
                          const int*    __restrict__ x,
                          int t,
                          float* __restrict__ cState,        // [64][512]
                          __bf16* __restrict__ hOutA,        // dest1 (pre-offset, stride 1024)
                          __bf16* __restrict__ hOutB,        // dest2 or null
                          float*  __restrict__ hOutF) {      // fp32 dest or null
  __shared__ float g_s[64][68];

  const int wg   = blockIdx.x;       // 0..31
  const int tid  = threadIdx.x;
  const int wave = tid >> 5;         // 0..7
  const int lane = tid & 31;
  const int nsub = wave & 3;
  const int mh   = wave >> 2;        // M half: rows [mh*32, mh*32+32)
  const int rsel = lane & 15;
  const int hsel = lane >> 4;        // 0/1
  const int ntile = wg*4 + nsub;     // 0..127

  v8f acc0 = {0.f,0.f,0.f,0.f,0.f,0.f,0.f,0.f};
  v8f acc1 = acc0;

  const __bf16* wptr = wp + ((size_t)ntile*32*32 + lane)*16;  // + kt*512 per kt
  const __bf16* a0 = aIn + (size_t)(mh*32 +  0 + rsel)*KDIM;
  const __bf16* a1 = aIn + (size_t)(mh*32 + 16 + rsel)*KDIM;

  // software pipeline: fragments for kt in flight while computing kt-1
  int k0 = hsel*8;
  v16bf bf  = *(const v16bf*)(wptr);
  v16bf af0 = ldA16(a0 + k0);
  v16bf af1 = ldA16(a1 + k0);
  #pragma unroll 4
  for (int ktn = 1; ktn < 32; ++ktn) {
    int k0n = ktn*32 + hsel*8;
    v16bf bf_n  = *(const v16bf*)(wptr + (size_t)ktn*512);
    v16bf af0_n = ldA16(a0 + k0n);
    v16bf af1_n = ldA16(a1 + k0n);
    acc0 = wmma_bf16(af0, bf, acc0);
    acc1 = wmma_bf16(af1, bf, acc1);
    bf = bf_n; af0 = af0_n; af1 = af1_n;
  }
  acc0 = wmma_bf16(af0, bf, acc0);
  acc1 = wmma_bf16(af1, bf, acc1);

  // D layout: lane holds column n'=ntile*16+rsel, rows M = r + 8*hsel (VGPR r)
  const int col = nsub*16 + rsel;     // n'_local in [0,64)
  const int mb  = mh*32 + hsel*8;
  #pragma unroll
  for (int r = 0; r < 8; ++r) {
    g_s[mb +  0 + r][col] = acc0[r];
    g_s[mb + 16 + r][col] = acc1[r];
  }
  __syncthreads();

  // Gate phase: 64 batch x 16 local-s = 1024 items, 4 per thread.
  #pragma unroll
  for (int it = 0; it < 4; ++it) {
    int idx = tid + it*256;
    int b  = idx >> 4;
    int sl = idx & 15;
    int s  = wg*16 + sl;
    float gi = g_s[b][sl*4 + 0];
    float gf = g_s[b][sl*4 + 1];
    float gg = g_s[b][sl*4 + 2];
    float go = g_s[b][sl*4 + 3];
    gi += bih[s]        + bhh[s];
    gf += bih[512 + s]  + bhh[512 + s];
    gg += bih[1024 + s] + bhh[1024 + s];
    go += bih[1536 + s] + bhh[1536 + s];
    if (gatherW) {                       // one-hot input column of Wih0
      int cls = x[b*NSTEP + t];
      gi += gatherW[(size_t)(s)       *576 + cls];
      gf += gatherW[(size_t)(512 + s) *576 + cls];
      gg += gatherW[(size_t)(1024 + s)*576 + cls];
      go += gatherW[(size_t)(1536 + s)*576 + cls];
    }
    float c  = cState[b*S_DIM + s];
    float cn = sigf(gf)*c + sigf(gi)*tanhf(gg);
    float hn = sigf(go)*tanhf(cn);
    cState[b*S_DIM + s] = cn;
    hOutA[b*KDIM + s] = (__bf16)hn;
    if (hOutB) hOutB[b*KDIM + s] = (__bf16)hn;
    if (hOutF) hOutF[b*S_DIM + s] = hn;
  }
}

// ---------------------------------------------------------------------------
// Attention + output: phi = Wphi@h2+bphi; e = phi.psi; softmax; ctx = a@h;
// p = Wcd@[h2|ctx]+bcd.  One workgroup per batch row.
// ---------------------------------------------------------------------------
__global__ void attn_step(const float* __restrict__ h2f,
                          const float* __restrict__ Wphi,
                          const float* __restrict__ bphi,
                          const float* __restrict__ psi,
                          const float* __restrict__ h,
                          const float* __restrict__ Wcd,
                          const float* __restrict__ bcd,
                          __bf16* __restrict__ ctxOut,       // A0[1-p] ctx half
                          float*  __restrict__ out,
                          int t) {
  __shared__ float h2s[512];
  __shared__ float phis[128];
  __shared__ float sc[512];
  __shared__ float red[256];
  __shared__ float cctxs[512];

  const int b = blockIdx.x;
  const int tid = threadIdx.x;

  h2s[tid]       = h2f[b*S_DIM + tid];
  h2s[tid + 256] = h2f[b*S_DIM + tid + 256];
  __syncthreads();

  // phi_s (128 outputs, 2-way K-split)
  {
    int o = tid & 127, half = tid >> 7, kb = half*256;
    const float* wr = Wphi + (size_t)o*512 + kb;
    float acc = 0.f;
    #pragma unroll 4
    for (int k = 0; k < 256; ++k) acc += wr[k]*h2s[kb + k];
    red[tid] = acc;
  }
  __syncthreads();
  if (tid < 128) phis[tid] = red[tid] + red[tid + 128] + bphi[tid];
  __syncthreads();

  // scores over T=512
  for (int tt = tid; tt < 512; tt += 256) {
    const float* pr = psi + ((size_t)b*TT + tt)*ATTN_D;
    float e = 0.f;
    #pragma unroll 4
    for (int a = 0; a < 128; ++a) e += phis[a]*pr[a];
    sc[tt] = e;
  }
  __syncthreads();

  // softmax (max, exp, sum)
  float m = fmaxf(sc[tid], sc[tid + 256]);
  red[tid] = m; __syncthreads();
  for (int s2 = 128; s2 > 0; s2 >>= 1) { if (tid < s2) red[tid] = fmaxf(red[tid], red[tid+s2]); __syncthreads(); }
  m = red[0]; __syncthreads();
  float e0 = __expf(sc[tid] - m), e1 = __expf(sc[tid + 256] - m);
  red[tid] = e0 + e1; __syncthreads();
  for (int s2 = 128; s2 > 0; s2 >>= 1) { if (tid < s2) red[tid] += red[tid+s2]; __syncthreads(); }
  float inv = 1.0f/red[0]; __syncthreads();
  sc[tid] = e0*inv; sc[tid + 256] = e1*inv;
  __syncthreads();

  // context = alpha @ h[b]   (each thread owns 2 columns of 512)
  {
    const float* hb = h + (size_t)b*TT*H2D;
    float c0 = 0.f, c1 = 0.f;
    for (int tt = 0; tt < 512; ++tt) {
      float al = sc[tt];
      const float* hr = hb + (size_t)tt*H2D;
      c0 += al*hr[tid];
      c1 += al*hr[tid + 256];
    }
    cctxs[tid] = c0;        cctxs[tid + 256] = c1;
    ctxOut[b*KDIM + tid]       = (__bf16)c0;
    ctxOut[b*KDIM + tid + 256] = (__bf16)c1;
  }
  __syncthreads();

  // p = Wcd @ [h2 | ctx] + bcd   (64 outputs, 4-way K-split)
  {
    int o = tid & 63, q = tid >> 6, kb = q*256;
    const float* wr = Wcd + (size_t)o*1024;
    float acc = 0.f;
    #pragma unroll 4
    for (int k = 0; k < 256; ++k) {
      int K = kb + k;
      float av = (K < 512) ? h2s[K] : cctxs[K - 512];
      acc += wr[K]*av;
    }
    red[tid] = acc;
  }
  __syncthreads();
  if (tid < 64)
    out[((size_t)b*NSTEP + t)*NCLS + tid] =
        red[tid] + red[tid + 64] + red[tid + 128] + red[tid + 192] + bcd[tid];
}

// ---------------------------------------------------------------------------
extern "C" void kernel_launch(void* const* d_in, const int* in_sizes, int n_in,
                              void* d_out, int out_size, void* d_ws, size_t ws_size,
                              hipStream_t stream) {
  (void)in_sizes; (void)n_in; (void)out_size; (void)ws_size;
  const int*   x    = (const int*)  d_in[0];
  const float* h    = (const float*)d_in[1];
  const float* Wih0 = (const float*)d_in[2];
  const float* Whh0 = (const float*)d_in[3];
  const float* bih0 = (const float*)d_in[4];
  const float* bhh0 = (const float*)d_in[5];
  const float* Wih1 = (const float*)d_in[6];
  const float* Whh1 = (const float*)d_in[7];
  const float* bih1 = (const float*)d_in[8];
  const float* bhh1 = (const float*)d_in[9];
  const float* Wphi = (const float*)d_in[10];
  const float* bphi = (const float*)d_in[11];
  const float* Wpsi = (const float*)d_in[12];
  const float* bpsi = (const float*)d_in[13];
  const float* Wcd  = (const float*)d_in[14];
  const float* bcd  = (const float*)d_in[15];
  float* out = (float*)d_out;

  char* ws = (char*)d_ws;
  size_t off = 0;
  auto take = [&](size_t bytes) { char* p = ws + off; off += (bytes + 255) & ~(size_t)255; return p; };
  __bf16* W0p = (__bf16*)take((size_t)KDIM*NDIM*2);
  __bf16* W1p = (__bf16*)take((size_t)KDIM*NDIM*2);
  __bf16* A0  = (__bf16*)take((size_t)2*BB*KDIM*2);   // ping-pong
  __bf16* A1  = (__bf16*)take((size_t)2*BB*KDIM*2);
  float*  c1  = (float*) take((size_t)BB*S_DIM*4);
  float*  c2  = (float*) take((size_t)BB*S_DIM*4);
  float*  h2f = (float*) take((size_t)BB*S_DIM*4);
  float*  psi = (float*) take((size_t)BB*TT*ATTN_D*4);

  pack_weights<<<8192, 256, 0, stream>>>(Wih0, Whh0, W0p, 576, 64);
  pack_weights<<<8192, 256, 0, stream>>>(Wih1, Whh1, W1p, 512, 0);
  init_state<<<256, 256, 0, stream>>>(h, A0, A1, c1, c2);
  psi_precompute<<<BB*TT/2, 256, 0, stream>>>(h, Wpsi, bpsi, psi);

  for (int t = 0; t < NSTEP; ++t) {
    int p = t & 1;
    __bf16* A0r = A0 + (size_t)p      * BB*KDIM;
    __bf16* A0w = A0 + (size_t)(1-p)  * BB*KDIM;
    __bf16* A1r = A1 + (size_t)p      * BB*KDIM;
    __bf16* A1w = A1 + (size_t)(1-p)  * BB*KDIM;
    // layer 0: reads A0[p]=[ctx|h1prev]; writes h1 -> A1[p][:,0:512] and A0[1-p][:,512:]
    lstm_step<<<32, 256, 0, stream>>>(A0r, W0p, bih0, bhh0, Wih0, x, t, c1,
                                      A1r, A0w + 512, nullptr);
    // layer 1: reads A1[p]=[h1|h2prev]; writes h2 -> A1[1-p][:,512:] and h2f
    lstm_step<<<32, 256, 0, stream>>>(A1r, W1p, bih1, bhh1, nullptr, x, t, c2,
                                      A1w + 512, nullptr, h2f);
    // attention + output; writes ctx -> A0[1-p][:,0:512]
    attn_step<<<BB, 256, 0, stream>>>(h2f, Wphi, bphi, psi, h, Wcd, bcd,
                                      A0w, out, t);
  }
}
